// DensePredictor_44169443672113
// MI455X (gfx1250) — matrix operand order repaired and checked by reading, not compiled, not run on hardware
//
#include <hip/hip_runtime.h>
#include <hip/hip_bf16.h>

typedef __attribute__((ext_vector_type(16))) _Float16 v16h;
typedef __attribute__((ext_vector_type(8)))  _Float16 v8h;
typedef __attribute__((ext_vector_type(8)))  float    v8f;

#define NB 8
#define NPTS 2048

// ---------------------------------------------------------------------------
// Utility kernels
// ---------------------------------------------------------------------------

__global__ void k_transpose(const float* __restrict__ in, float* __restrict__ out,
                            int B, int N) {
  int t = blockIdx.x * blockDim.x + threadIdx.x;
  if (t >= B * N) return;
  int b = t / N, n = t % N;
  out[(size_t)t * 3 + 0] = in[((size_t)b * 3 + 0) * N + n];
  out[(size_t)t * 3 + 1] = in[((size_t)b * 3 + 1) * N + n];
  out[(size_t)t * 3 + 2] = in[((size_t)b * 3 + 2) * N + n];
}

__global__ void k_inv_density(const float* __restrict__ xyz, float* __restrict__ invd,
                              int B, int N, float bw) {
  int t = blockIdx.x * blockDim.x + threadIdx.x;
  if (t >= B * N) return;
  int b = t / N, i = t % N;
  const float* base = xyz + (size_t)b * N * 3;
  float xi = base[i * 3 + 0], yi = base[i * 3 + 1], zi = base[i * 3 + 2];
  float inv2bw2 = 1.0f / (2.0f * bw * bw);
  float s = 0.0f;
  for (int j = 0; j < N; ++j) {
    float dx = base[j * 3 + 0] - xi;
    float dy = base[j * 3 + 1] - yi;
    float dz = base[j * 3 + 2] - zi;
    float d = dx * dx + dy * dy + dz * dz;
    s += expf(-d * inv2bw2);
  }
  float dens = (s / (2.5f * bw)) / (float)N;
  invd[t] = 1.0f / dens;
}

// Farthest point sampling: one block per batch, sequential scan as in reference.
__global__ void k_fps(const float* __restrict__ xyz, int* __restrict__ outIdx,
                      int N, int npoint) {
  __shared__ float sx[NPTS * 3];
  __shared__ float sd[NPTS];
  __shared__ float rv[256];
  __shared__ int   ri[256];
  __shared__ int   sfar;
  int b = blockIdx.x;
  const float* base = xyz + (size_t)b * N * 3;
  for (int i = threadIdx.x; i < N * 3; i += blockDim.x) sx[i] = base[i];
  for (int i = threadIdx.x; i < N; i += blockDim.x) sd[i] = 1e10f;
  if (threadIdx.x == 0) sfar = 0;
  __syncthreads();
  for (int it = 0; it < npoint; ++it) {
    int far = sfar;
    if (threadIdx.x == 0) outIdx[b * npoint + it] = far;
    float fx = sx[far * 3 + 0], fy = sx[far * 3 + 1], fz = sx[far * 3 + 2];
    float lmax = -1.0f; int lidx = 0;
    for (int i = threadIdx.x; i < N; i += blockDim.x) {
      float dx = sx[i * 3 + 0] - fx, dy = sx[i * 3 + 1] - fy, dz = sx[i * 3 + 2] - fz;
      float d = dx * dx + dy * dy + dz * dz;
      float nd = fminf(sd[i], d);
      sd[i] = nd;
      if (nd > lmax) { lmax = nd; lidx = i; }
    }
    rv[threadIdx.x] = lmax; ri[threadIdx.x] = lidx;
    __syncthreads();
    for (int s = 128; s > 0; s >>= 1) {
      if (threadIdx.x < s) {
        if (rv[threadIdx.x + s] > rv[threadIdx.x]) {
          rv[threadIdx.x] = rv[threadIdx.x + s];
          ri[threadIdx.x] = ri[threadIdx.x + s];
        }
      }
      __syncthreads();
    }
    if (threadIdx.x == 0) sfar = ri[0];
    __syncthreads();
  }
}

__global__ void k_gather_xyz(const float* __restrict__ xyz, const int* __restrict__ idx,
                             float* __restrict__ out, int B, int N, int S) {
  int t = blockIdx.x * blockDim.x + threadIdx.x;
  if (t >= B * S) return;
  int b = t / S;
  int j = idx[t];
  out[(size_t)t * 3 + 0] = xyz[((size_t)b * N + j) * 3 + 0];
  out[(size_t)t * 3 + 1] = xyz[((size_t)b * N + j) * 3 + 1];
  out[(size_t)t * 3 + 2] = xyz[((size_t)b * N + j) * 3 + 2];
}

// kNN via iterative min extraction (sorted ascending, matches top_k(-sqdist)).
__global__ void k_knn(const float* __restrict__ xyz, const float* __restrict__ cent,
                      int* __restrict__ outIdx, int B, int N, int S, int K) {
  __shared__ float sd[NPTS];
  __shared__ float rv[256];
  __shared__ int   ri[256];
  int bs = blockIdx.x;
  int b = bs / S;
  const float* base = xyz + (size_t)b * N * 3;
  float cx = cent[(size_t)bs * 3 + 0], cy = cent[(size_t)bs * 3 + 1], cz = cent[(size_t)bs * 3 + 2];
  for (int i = threadIdx.x; i < N; i += blockDim.x) {
    float dx = base[i * 3 + 0] - cx, dy = base[i * 3 + 1] - cy, dz = base[i * 3 + 2] - cz;
    sd[i] = dx * dx + dy * dy + dz * dz;
  }
  __syncthreads();
  for (int k = 0; k < K; ++k) {
    float lmin = 1e30f; int lidx = 0;
    for (int i = threadIdx.x; i < N; i += blockDim.x)
      if (sd[i] < lmin) { lmin = sd[i]; lidx = i; }
    rv[threadIdx.x] = lmin; ri[threadIdx.x] = lidx;
    __syncthreads();
    for (int s = 128; s > 0; s >>= 1) {
      if (threadIdx.x < s) {
        if (rv[threadIdx.x + s] < rv[threadIdx.x]) {
          rv[threadIdx.x] = rv[threadIdx.x + s];
          ri[threadIdx.x] = ri[threadIdx.x + s];
        }
      }
      __syncthreads();
    }
    if (threadIdx.x == 0) { outIdx[(size_t)bs * K + k] = ri[0]; sd[ri[0]] = 1e30f; }
    __syncthreads();
  }
}

// Build grouped features: newpts (BSK, 3+C) = [gxyz | gathered pts], gdens (BSK)
__global__ void k_group(const float* __restrict__ xyz, const float* __restrict__ pts,
                        const float* __restrict__ cent, const int* __restrict__ idx,
                        const float* __restrict__ invd,
                        float* __restrict__ newpts, float* __restrict__ gdens,
                        int B, int N, int S, int K, int C) {
  int t = blockIdx.x * blockDim.x + threadIdx.x;
  int total = B * S * K;
  if (t >= total) return;
  int bs = t / K;
  int b = bs / S;
  int j = idx[t];
  const float* p = xyz + ((size_t)b * N + j) * 3;
  const float* c = cent + (size_t)bs * 3;
  float* o = newpts + (size_t)t * (3 + C);
  o[0] = p[0] - c[0]; o[1] = p[1] - c[1]; o[2] = p[2] - c[2];
  const float* f = pts + ((size_t)b * N + j) * C;
  for (int cc = 0; cc < C; ++cc) o[3 + cc] = f[cc];
  gdens[t] = invd[(size_t)b * N + j];
}

// group_all: gxyz = raw coords (uncentered), all N points, S=1.
__global__ void k_group_all(const float* __restrict__ xyz, const float* __restrict__ pts,
                            const float* __restrict__ invd,
                            float* __restrict__ newpts, float* __restrict__ gdens,
                            int B, int N, int C) {
  int t = blockIdx.x * blockDim.x + threadIdx.x;
  if (t >= B * N) return;
  float* o = newpts + (size_t)t * (3 + C);
  o[0] = xyz[(size_t)t * 3 + 0];
  o[1] = xyz[(size_t)t * 3 + 1];
  o[2] = xyz[(size_t)t * 3 + 2];
  const float* f = pts + (size_t)t * C;
  for (int cc = 0; cc < C; ++cc) o[3 + cc] = f[cc];
  gdens[t] = invd[t];
}

__global__ void k_gdens_norm(float* __restrict__ gdens, int BS, int K) {
  int t = blockIdx.x * blockDim.x + threadIdx.x;
  if (t >= BS) return;
  float* g = gdens + (size_t)t * K;
  float m = -1e30f;
  for (int k = 0; k < K; ++k) m = fmaxf(m, g[k]);
  for (int k = 0; k < K; ++k) g[k] /= m;
}

// ---------------------------------------------------------------------------
// WMMA GEMM: Out[M,N] = A[M,K] * W[N,K]^T + bias.  f32 in/out, f16 WMMA core.
// Block tile 64x32 (8 waves, each wave = one 16x16 tile).
// A tile row-major in LDS; B tile stored TRANSPOSED (n-major) so both
// fragments load as contiguous 128-bit LDS reads (ds_load_b128).
// ---------------------------------------------------------------------------

__global__ void __launch_bounds__(256)
k_gemm_wmma(const float* __restrict__ A, int lda,
            const float* __restrict__ W, const float* __restrict__ bias,
            float* __restrict__ Out, int M, int Nn, int Kk) {
  __shared__ __align__(16) _Float16 sA[64 * 32];   // [row][k]
  __shared__ __align__(16) _Float16 sBt[32 * 32];  // [n][k]  (transposed)
  int tid = threadIdx.x;
  int lane = tid & 31, wave = tid >> 5;
  int wm = wave & 3, wn = wave >> 2;
  int bm = blockIdx.x * 64;
  int bn = blockIdx.y * 32;
  int r = lane & 15, half = lane >> 4;
  int mrow = wm * 16 + r;
  int ncol = wn * 16 + r;
  v8f acc = {};
  for (int k0 = 0; k0 < Kk; k0 += 32) {
    // stage A tile (64 x 32) as f16
    for (int i = tid; i < 64 * 32; i += 256) {
      int row = i >> 5, col = i & 31;
      int gm = bm + row, gk = k0 + col;
      float v = (gm < M && gk < Kk) ? A[(size_t)gm * lda + gk] : 0.0f;
      sA[row * 32 + col] = (_Float16)v;
    }
    // stage B tile transposed: sBt[n][k] = W[bn+n][k0+k]
    for (int i = tid; i < 32 * 32; i += 256) {
      int nn = i >> 5, kk = i & 31;
      int gn = bn + nn, gk = k0 + kk;
      float v = (gn < Nn && gk < Kk) ? W[(size_t)gn * Kk + gk] : 0.0f;
      sBt[nn * 32 + kk] = (_Float16)v;
    }
    // speculative prefetch of next A K-tile (global_prefetch_b8)
    if (k0 + 32 < Kk) {
      int prow = bm + (tid >> 2);
      if (prow < M)
        __builtin_prefetch(&A[(size_t)prow * lda + k0 + 32 + (tid & 3) * 8], 0, 0);
    }
    __syncthreads();
    // A fragment: two contiguous 8-half runs per lane -> 2x ds_load_b128
    v8h a0 = *(const v8h*)&sA[mrow * 32 + 8 * half];
    v8h a1 = *(const v8h*)&sA[mrow * 32 + 16 + 8 * half];
    // B fragment: two contiguous 8-half runs per lane -> 2x ds_load_b128
    v8h b0 = *(const v8h*)&sBt[ncol * 32 + 8 * half];
    v8h b1 = *(const v8h*)&sBt[ncol * 32 + 16 + 8 * half];
    v16h af, bf;
#pragma unroll
    for (int i = 0; i < 8; ++i) {
      af[i] = a0[i]; af[i + 8] = a1[i];
      bf[i] = b0[i]; bf[i + 8] = b1[i];
    }
    acc = __builtin_amdgcn_wmma_f32_16x16x32_f16(false, af, false, bf,
                                                 (short)0, acc, false, false);
    __syncthreads();
  }
#pragma unroll
  for (int j = 0; j < 8; ++j) {
    int gm = bm + wm * 16 + j + 8 * half;
    int gn = bn + wn * 16 + r;
    if (gm < M && gn < Nn) {
      float v = acc[j];
      if (bias) v += bias[gn];
      Out[(size_t)gm * Nn + gn] = v;
    }
  }
}

// einsum 'bskc,bskw->bscw' with w-dim = 16: one wave per (bs, c-tile).
__global__ void k_einsum_wmma(const float* __restrict__ feat, const float* __restrict__ wn,
                              float* __restrict__ agg, int BS, int K, int C) {
  int tile = blockIdx.x;
  int ctiles = C >> 4;
  int ct = tile % ctiles;
  int bs = tile / ctiles;
  int lane = threadIdx.x;
  int r = lane & 15, half = lane >> 4;
  const float* fb = feat + (size_t)bs * K * C;
  const float* wb = wn + (size_t)bs * K * 16;
  int m = ct * 16 + r;
  v8f acc = {};
  for (int k0 = 0; k0 < K; k0 += 32) {
    v16h af, bf;
#pragma unroll
    for (int i = 0; i < 16; ++i) {
      int ka = k0 + (i < 8 ? i : i + 8) + 8 * half;
      af[i] = (_Float16)fb[(size_t)ka * C + m];
      int kb = k0 + (i < 8 ? i : i + 8) + 8 * half;
      bf[i] = (_Float16)wb[(size_t)kb * 16 + r];
    }
    acc = __builtin_amdgcn_wmma_f32_16x16x32_f16(false, af, false, bf,
                                                 (short)0, acc, false, false);
  }
#pragma unroll
  for (int j = 0; j < 8; ++j) {
    int c = ct * 16 + j + 8 * half;
    agg[((size_t)bs * C + c) * 16 + r] = acc[j];
  }
}

// ---------------------------------------------------------------------------
// GroupNorm (reduces over all non-batch axes) + affine + ReLU
// ---------------------------------------------------------------------------

__global__ void k_meanvar(const float* __restrict__ X, float* __restrict__ mv,
                          size_t perBatch) {
  __shared__ float s1[256], s2[256];
  int b = blockIdx.x;
  const float* base = X + (size_t)b * perBatch;
  float a = 0.0f, q = 0.0f;
  for (size_t i = threadIdx.x; i < perBatch; i += blockDim.x) {
    float v = base[i]; a += v; q += v * v;
  }
  s1[threadIdx.x] = a; s2[threadIdx.x] = q;
  __syncthreads();
  for (int st = 128; st > 0; st >>= 1) {
    if (threadIdx.x < st) { s1[threadIdx.x] += s1[threadIdx.x + st]; s2[threadIdx.x] += s2[threadIdx.x + st]; }
    __syncthreads();
  }
  if (threadIdx.x == 0) {
    float m = s1[0] / (float)perBatch;
    float v = s2[0] / (float)perBatch - m * m;
    mv[b * 2 + 0] = m;
    mv[b * 2 + 1] = v;
  }
}

__global__ void k_gn_affine(const float* __restrict__ X, float* __restrict__ Y,
                            const float* __restrict__ mv, const float* __restrict__ g,
                            const float* __restrict__ beta,
                            size_t total, size_t rowsPerBatch, int C, int relu) {
  size_t t = (size_t)blockIdx.x * blockDim.x + threadIdx.x;
  if (t >= total) return;
  int c = (int)(t % C);
  int b = (int)(t / (rowsPerBatch * (size_t)C));
  float m = mv[b * 2 + 0], v = mv[b * 2 + 1];
  float y = (X[t] - m) * rsqrtf(v + 1e-5f) * g[c] + beta[c];
  if (relu) y = fmaxf(y, 0.0f);
  Y[t] = y;
}

__global__ void k_mul_scale(float* __restrict__ feat, const float* __restrict__ scale,
                            size_t rows, int C) {
  size_t t = (size_t)blockIdx.x * blockDim.x + threadIdx.x;
  if (t >= rows * (size_t)C) return;
  feat[t] *= scale[t / C];
}

// ---------------------------------------------------------------------------
// Feature propagation: 3-NN inverse-distance interpolation
// ---------------------------------------------------------------------------

__global__ void k_interp3(const float* __restrict__ xyz1, const float* __restrict__ xyz2,
                          const float* __restrict__ pts2, float* __restrict__ out,
                          int B, int N1, int N2, int C) {
  int t = blockIdx.x * blockDim.x + threadIdx.x;
  if (t >= B * N1) return;
  int b = t / N1;
  const float* q = xyz1 + (size_t)t * 3;
  const float* base = xyz2 + (size_t)b * N2 * 3;
  float d0 = 1e30f, d1 = 1e30f, d2 = 1e30f;
  int i0 = 0, i1 = 0, i2 = 0;
  for (int j = 0; j < N2; ++j) {
    float dx = base[j * 3 + 0] - q[0];
    float dy = base[j * 3 + 1] - q[1];
    float dz = base[j * 3 + 2] - q[2];
    float d = dx * dx + dy * dy + dz * dz;
    if (d < d0)      { d2 = d1; i2 = i1; d1 = d0; i1 = i0; d0 = d; i0 = j; }
    else if (d < d1) { d2 = d1; i2 = i1; d1 = d;  i1 = j; }
    else if (d < d2) { d2 = d;  i2 = j; }
  }
  float w0 = 1.0f / (d0 + 1e-8f), w1 = 1.0f / (d1 + 1e-8f), w2 = 1.0f / (d2 + 1e-8f);
  float ws = w0 + w1 + w2;
  w0 /= ws; w1 /= ws; w2 /= ws;
  const float* p = pts2 + (size_t)b * N2 * C;
  float* o = out + (size_t)t * C;
  for (int c = 0; c < C; ++c)
    o[c] = w0 * p[(size_t)i0 * C + c] + w1 * p[(size_t)i1 * C + c] + w2 * p[(size_t)i2 * C + c];
}

__global__ void k_concat(const float* __restrict__ A, const float* __restrict__ Bx,
                         float* __restrict__ out, size_t rows, int C1, int C2,
                         int bBcast, size_t rowsPerBatch) {
  size_t t = (size_t)blockIdx.x * blockDim.x + threadIdx.x;
  int CT = C1 + C2;
  if (t >= rows * (size_t)CT) return;
  size_t row = t / CT;
  int c = (int)(t % CT);
  if (c < C1) out[t] = A[row * C1 + c];
  else {
    size_t brow = bBcast ? (row / rowsPerBatch) : row;
    out[t] = Bx[brow * C2 + (c - C1)];
  }
}

__global__ void k_logsoftmax2(const float* __restrict__ logits, float* __restrict__ out,
                              int B, int N) {
  int t = blockIdx.x * blockDim.x + threadIdx.x;
  if (t >= B * N) return;
  int b = t / N, n = t % N;
  float l0 = logits[(size_t)t * 2 + 0], l1 = logits[(size_t)t * 2 + 1];
  float m = fmaxf(l0, l1);
  float lse = m + logf(expf(l0 - m) + expf(l1 - m));
  out[((size_t)b * 2 + 0) * N + n] = l0 - lse;
  out[((size_t)b * 2 + 1) * N + n] = l1 - lse;
}

// ---------------------------------------------------------------------------
// Host-side orchestration
// ---------------------------------------------------------------------------

struct Layer { const float *W, *b, *g, *beta; };
struct SA {
  Layer mlp;
  Layer wn[3];
  Layer dn[3];
  const float *linW, *linB, *bnG, *bnBeta;
};
struct Scratch {
  float *invd, *fidx, *idx, *newpts, *gdens, *feat, *sc1, *sc2, *agg, *mv;
};
struct Persist {
  float *xyz, *l1x, *l1p, *l2x, *l2p, *l3p, *fout;
};

static inline void gemm(const float* A, int lda, const float* W, const float* bias,
                        float* Out, int M, int Nn, int Kk, hipStream_t st) {
  dim3 grid((unsigned)((M + 63) / 64), (unsigned)((Nn + 31) / 32));
  k_gemm_wmma<<<grid, 256, 0, st>>>(A, lda, W, bias, Out, M, Nn, Kk);
}

static void pw_gn_relu(const float* A, int lda, const Layer& L, float* Y, float* mv,
                       size_t rowsPerBatch, int Kk, int Nc, int relu, hipStream_t st) {
  size_t M = (size_t)NB * rowsPerBatch;
  gemm(A, lda, L.W, L.b, Y, (int)M, Nc, Kk, st);
  k_meanvar<<<NB, 256, 0, st>>>(Y, mv, rowsPerBatch * (size_t)Nc);
  size_t total = M * (size_t)Nc;
  k_gn_affine<<<(unsigned)((total + 255) / 256), 256, 0, st>>>(
      Y, Y, mv, L.g, L.beta, total, rowsPerBatch, Nc, relu);
}

static void sa_common_tail(const SA& P, Scratch& sc, float* outp,
                           size_t SK, size_t BSK, int BS, int Cin, int K, int Cout,
                           hipStream_t st) {
  // feat = relu(GN(pw(newpts, mlp)))
  pw_gn_relu(sc.newpts, Cin, P.mlp, sc.feat, sc.mv, SK, Cin, Cout, 1, st);
  // scale = density MLP on normalized inverse density
  pw_gn_relu(sc.gdens, 1, P.dn[0], sc.sc1, sc.mv, SK, 1, 16, 1, st);
  pw_gn_relu(sc.sc1, 16, P.dn[1], sc.sc2, sc.mv, SK, 16, 8, 1, st);
  pw_gn_relu(sc.sc2, 8, P.dn[2], sc.gdens, sc.mv, SK, 8, 1, 1, st);
  k_mul_scale<<<(unsigned)((BSK * Cout + 255) / 256), 256, 0, st>>>(sc.feat, sc.gdens, BSK, Cout);
  // weightnet on gxyz (first 3 columns of newpts, strided lda)
  pw_gn_relu(sc.newpts, Cin, P.wn[0], sc.sc1, sc.mv, SK, 3, 8, 1, st);
  pw_gn_relu(sc.sc1, 8, P.wn[1], sc.sc2, sc.mv, SK, 8, 8, 1, st);
  pw_gn_relu(sc.sc2, 8, P.wn[2], sc.sc1, sc.mv, SK, 8, 16, 1, st);
  // einsum + linear + GN + relu
  k_einsum_wmma<<<(unsigned)(BS * (Cout / 16)), 32, 0, st>>>(sc.feat, sc.sc1, sc.agg, BS, K, Cout);
  Layer LL; LL.W = P.linW; LL.b = P.linB; LL.g = P.bnG; LL.beta = P.bnBeta;
  pw_gn_relu(sc.agg, Cout * 16, LL, outp, sc.mv, SK / (size_t)K, Cout * 16, Cout, 1, st);
}

static void sa_forward(const SA& P, const float* pxyz, const float* ppts,
                       int Nin, int Cpts, int S, int K, float bw, int Cout,
                       float* nxyz, float* outp, Scratch& sc, hipStream_t st) {
  int Cin = 3 + Cpts;
  int BS = NB * S;
  size_t BSK = (size_t)BS * K;
  k_inv_density<<<(unsigned)((NB * Nin + 255) / 256), 256, 0, st>>>(pxyz, sc.invd, NB, Nin, bw);
  k_fps<<<NB, 256, 0, st>>>(pxyz, (int*)sc.fidx, Nin, S);
  k_gather_xyz<<<(unsigned)((BS + 255) / 256), 256, 0, st>>>(pxyz, (const int*)sc.fidx, nxyz, NB, Nin, S);
  k_knn<<<(unsigned)BS, 256, 0, st>>>(pxyz, nxyz, (int*)sc.idx, NB, Nin, S, K);
  k_group<<<(unsigned)((BSK + 255) / 256), 256, 0, st>>>(pxyz, ppts, nxyz, (const int*)sc.idx,
                                                         sc.invd, sc.newpts, sc.gdens,
                                                         NB, Nin, S, K, Cpts);
  k_gdens_norm<<<(unsigned)((BS + 255) / 256), 256, 0, st>>>(sc.gdens, BS, K);
  sa_common_tail(P, sc, outp, (size_t)S * K, BSK, BS, Cin, K, Cout, st);
}

static void sa3_forward(const SA& P, const float* pxyz, const float* ppts,
                        int Nin, int Cpts, int Cout, float* outp, Scratch& sc,
                        hipStream_t st) {
  int Cin = 3 + Cpts;
  int K = Nin;
  size_t BSK = (size_t)NB * K;
  k_inv_density<<<(unsigned)((NB * Nin + 255) / 256), 256, 0, st>>>(pxyz, sc.invd, NB, Nin, 0.4f);
  k_group_all<<<(unsigned)((BSK + 255) / 256), 256, 0, st>>>(pxyz, ppts, sc.invd,
                                                             sc.newpts, sc.gdens, NB, Nin, Cpts);
  k_gdens_norm<<<1, 256, 0, st>>>(sc.gdens, NB, K);
  sa_common_tail(P, sc, outp, (size_t)K, BSK, NB, Cin, K, Cout, st);
}

static void run_branch(const float* in_b3n, Persist& P, Scratch& S,
                       const SA& sa1, const SA& sa2, const SA& sa3,
                       const Layer& fp3l, const Layer& fp2l,
                       const Layer& fp1a, const Layer& fp1b, hipStream_t st) {
  k_transpose<<<(NB * NPTS + 255) / 256, 256, 0, st>>>(in_b3n, P.xyz, NB, NPTS);
  sa_forward(sa1, P.xyz, P.xyz, NPTS, 3, 512, 32, 0.1f, 64, P.l1x, P.l1p, S, st);
  sa_forward(sa2, P.l1x, P.l1p, 512, 64, 128, 64, 0.2f, 128, P.l2x, P.l2p, S, st);
  sa3_forward(sa3, P.l2x, P.l2p, 128, 128, 256, P.l3p, S, st);
  // fp3: interp = broadcast(l3p); x = concat(l2p, interp) -> mlp 384->128
  {
    size_t rows = (size_t)NB * 128;
    size_t tot = rows * 384;
    k_concat<<<(unsigned)((tot + 255) / 256), 256, 0, st>>>(P.l2p, P.l3p, S.newpts, rows, 128, 256, 1, 128);
    pw_gn_relu(S.newpts, 384, fp3l, P.l2p, S.mv, 128, 384, 128, 1, st);
  }
  // fp2: 3-NN interp l2p -> 512 pts; x = concat(l1p, interp) -> mlp 192->64
  {
    k_interp3<<<(NB * 512 + 255) / 256, 256, 0, st>>>(P.l1x, P.l2x, P.l2p, S.feat, NB, 512, 128, 128);
    size_t rows = (size_t)NB * 512;
    size_t tot = rows * 192;
    k_concat<<<(unsigned)((tot + 255) / 256), 256, 0, st>>>(P.l1p, S.feat, S.newpts, rows, 64, 128, 0, 0);
    pw_gn_relu(S.newpts, 192, fp2l, P.l1p, S.mv, 512, 192, 64, 1, st);
  }
  // fp1: 3-NN interp l1p -> 2048 pts; mlp 64->64->64
  {
    k_interp3<<<(NB * NPTS + 255) / 256, 256, 0, st>>>(P.xyz, P.l1x, P.l1p, S.feat, NB, NPTS, 512, 64);
    pw_gn_relu(S.feat, 64, fp1a, S.sc1, S.mv, NPTS, 64, 64, 1, st);
    pw_gn_relu(S.sc1, 64, fp1b, P.fout, S.mv, NPTS, 64, 64, 1, st);
  }
}

extern "C" void kernel_launch(void* const* d_in, const int* in_sizes, int n_in,
                              void* d_out, int out_size, void* d_ws, size_t ws_size,
                              hipStream_t stream) {
  (void)in_sizes; (void)n_in; (void)out_size;

  // ---- unpack parameters in setup_inputs() dict order ----
  int pi = 2;
  auto nf = [&]() { return (const float*)d_in[pi++]; };
  auto nlayer = [&]() { Layer L; L.W = nf(); L.b = nf(); L.g = nf(); L.beta = nf(); return L; };
  auto nsa = [&]() {
    SA s;
    s.mlp = nlayer();
    for (int i = 0; i < 3; ++i) s.wn[i] = nlayer();
    for (int i = 0; i < 3; ++i) s.dn[i] = nlayer();
    s.linW = nf(); s.linB = nf(); s.bnG = nf(); s.bnBeta = nf();
    return s;
  };
  SA sa1 = nsa(), sa2 = nsa(), sa3 = nsa();
  Layer fp3l = nlayer();
  Layer fp2l = nlayer();
  Layer fp1a = nlayer();
  Layer fp1b = nlayer();
  const float* c1W = nf(); const float* c1b = nf();
  const float* g1g = nf(); const float* g1b = nf();
  const float* c2W = nf(); const float* c2b = nf();

  // ---- workspace layout (floats) ----
  float* w = (float*)d_ws;
  size_t off = 0;
  auto alloc = [&](size_t n) { float* p = w + off; off += n; return p; };

  Persist PA, PB;
  auto allocP = [&](Persist& P) {
    P.xyz  = alloc((size_t)NB * NPTS * 3);
    P.l1x  = alloc((size_t)NB * 512 * 3);
    P.l1p  = alloc((size_t)NB * 512 * 64);
    P.l2x  = alloc((size_t)NB * 128 * 3);
    P.l2p  = alloc((size_t)NB * 128 * 128);
    P.l3p  = alloc((size_t)NB * 256);
    P.fout = alloc((size_t)NB * NPTS * 64);
  };
  allocP(PA);
  allocP(PB);

  Scratch S;
  S.invd   = alloc((size_t)NB * NPTS);           // inverse density
  S.fidx   = alloc((size_t)NB * 512);            // FPS indices (int)
  S.idx    = alloc((size_t)NB * 512 * 32);       // kNN indices (int), max 131072
  S.newpts = alloc(4390912);                     // max: sa2 (65536 x 67)
  S.gdens  = alloc((size_t)NB * 512 * 32);       // grouped density / scale
  S.feat   = alloc(8388608);                     // max: sa1/sa2 feat
  S.sc1    = alloc(2097152);                     // max: BSK x 16
  S.sc2    = alloc(1048576);                     // max: BSK x 8
  S.agg    = alloc(4194304);                     // max: sa1 agg (4096 x 1024)
  S.mv     = alloc(2 * NB);                      // per-batch mean/var

  if (ws_size < off * sizeof(float)) return;     // insufficient workspace

  // ---- two shared-weight branches ----
  run_branch((const float*)d_in[0], PA, S, sa1, sa2, sa3, fp3l, fp2l, fp1a, fp1b, stream);
  run_branch((const float*)d_in[1], PB, S, sa1, sa2, sa3, fp3l, fp2l, fp1a, fp1b, stream);

  // ---- head: concat -> conv1+GN+relu -> conv2 -> log_softmax -> (B,2,N) ----
  {
    size_t rows = (size_t)NB * NPTS;
    size_t tot = rows * 128;
    k_concat<<<(unsigned)((tot + 255) / 256), 256, 0, stream>>>(PA.fout, PB.fout, S.newpts, rows, 64, 64, 0, 0);
    Layer h1; h1.W = c1W; h1.b = c1b; h1.g = g1g; h1.beta = g1b;
    pw_gn_relu(S.newpts, 128, h1, S.feat, S.mv, NPTS, 128, 64, 1, stream);
    gemm(S.feat, 64, c2W, c2b, S.sc1, NB * NPTS, 2, 64, stream);
    k_logsoftmax2<<<(NB * NPTS + 255) / 256, 256, 0, stream>>>(S.sc1, (float*)d_out, NB, NPTS);
  }
}
